// EfficientVectorQuantiser_25537875542500
// MI455X (gfx1250) — matrix operand order repaired and checked by reading, not compiled, not run on hardware
//
#include <hip/hip_runtime.h>
#include <hip/hip_bf16.h>
#include <math.h>

typedef _Float16 v16h __attribute__((ext_vector_type(16)));
typedef _Float16 v8h  __attribute__((ext_vector_type(8)));
typedef float    v8f  __attribute__((ext_vector_type(8)));

#define VQ_B   32
#define VQ_C   256
#define VQ_H   32
#define VQ_W   32
#define VQ_K   8192
#define VQ_N   (VQ_B * VQ_H * VQ_W)               // 32768
#define VQ_HW  (VQ_H * VQ_W)                      // 1024
#define VQ_OUT_ELEMS (VQ_B * VQ_C * VQ_H * VQ_W)  // 8388608

#define KSPLIT    8
#define TCODES    32                              // codes per LDS tile
#define KT_PER    (VQ_K / TCODES / KSPLIT)        // 32 tiles per block
#define ROWS_PB   128                             // rows per block (4 waves)
#define NBLK_ROWS (VQ_N / ROWS_PB)                // 256
#define LDSP      264                             // padded halfs per B row (528B)
#define TILE_PAD_HALFS (TCODES * LDSP)            // one padded tile (halfs)
#define TILE_PAD_BYTES (TCODES * 528)
#define TILE_SRC_BYTES (TCODES * VQ_C * 2)        // 16 KiB contiguous source

// ---------------------------------------------------------------------------
// Kernel 1: L2-normalize codebook rows -> f16.  One block (256 thr) per row.
// ---------------------------------------------------------------------------
__global__ void vq_norm_embed(const float* __restrict__ embed,
                              _Float16* __restrict__ cn) {
    const int k = blockIdx.x;
    const int t = threadIdx.x;
    __shared__ float red[VQ_C];
    __shared__ float invs;
    const float v = embed[(long)k * VQ_C + t];
    red[t] = v * v;
    __syncthreads();
    #pragma unroll
    for (int s = 128; s > 0; s >>= 1) {
        if (t < s) red[t] += red[t + s];
        __syncthreads();
    }
    if (t == 0) invs = 1.0f / fmaxf(sqrtf(red[0]), 1e-12f);
    __syncthreads();
    cn[(long)k * VQ_C + t] = (_Float16)(v * invs);
}

// ---------------------------------------------------------------------------
// Kernel 2: normalize z pixels, transpose to [N][C] row-major f16.
// ---------------------------------------------------------------------------
__global__ void vq_norm_z(const float* __restrict__ z,
                          _Float16* __restrict__ zn) {
    const int bh = blockIdx.x;
    const int b  = bh >> 5;
    const int h  = bh & 31;
    const int t  = threadIdx.x;
    const int w  = t & 31;
    const int cg = t >> 5;
    __shared__ float part[256];
    __shared__ float inv[32];
    const float* zb = z + (((long)b * VQ_C) * VQ_H + h) * VQ_W;
    float ss = 0.0f;
    #pragma unroll 4
    for (int j = 0; j < 32; ++j) {
        const int c = cg * 32 + j;
        const float v = zb[(long)c * VQ_HW + w];
        ss += v * v;
    }
    part[t] = ss;
    __syncthreads();
    if (t < 32) {
        float s = 0.0f;
        #pragma unroll
        for (int g = 0; g < 8; ++g) s += part[g * 32 + t];
        inv[t] = 1.0f / fmaxf(sqrtf(s), 1e-12f);
    }
    __syncthreads();
    const long nbase = (long)bh * VQ_W;
    #pragma unroll 4
    for (int cb = 0; cb < VQ_C; cb += 8) {
        const int c = cb + cg;
        const float v = zb[(long)c * VQ_HW + w];
        zn[(nbase + w) * VQ_C + c] = (_Float16)(v * inv[w]);
    }
}

// ---------------------------------------------------------------------------
// Kernel 3: fused cosine GEMM + argmax, K-split.
// 4 waves/block, each wave owns 32 rows (A register-resident).
// 32-code B tiles staged via global_load_async_to_lds_b128, TRIPLE-buffered:
// one barrier per tile (stage(kt+2) reuses the buffer of tile kt-1, which all
// waves finished before this barrier).  B fragments rotate one chunk ahead so
// ds_load latency overlaps the 4-WMMA group.
// ---------------------------------------------------------------------------
__global__ __launch_bounds__(128) void vq_dist_argmax(
        const _Float16* __restrict__ zn, const _Float16* __restrict__ cn,
        float* __restrict__ pbest, int* __restrict__ pidx) {
    __shared__ __align__(16) _Float16 smem[3 * TILE_PAD_HALFS];

    const int t      = threadIdx.x;
    const int lane   = t & 31;
    const int wave   = t >> 5;
    const int rowblk = blockIdx.x & (NBLK_ROWS - 1);
    const int ks     = blockIdx.x >> 8;                // K-slice 0..7
    const int mbase  = rowblk * ROWS_PB + wave * 32;
    const int cbase  = ks * (VQ_K / KSPLIT);           // first code of slice
    const int n15    = lane & 15;
    const int hi     = lane >> 4;

    // ---- preload A fragments: 2 row groups x 8 C-chunks, register resident
    v16h a0[8], a1[8];
    const _Float16* zr0 = zn + (long)(mbase + n15) * VQ_C;
    const _Float16* zr1 = zn + (long)(mbase + 16 + n15) * VQ_C;
    #pragma unroll
    for (int cc = 0; cc < 8; ++cc) {
        const int koff = cc * 32 + (hi ? 16 : 0);
        ((v8h*)&a0[cc])[0] = *(const v8h*)(zr0 + koff);
        ((v8h*)&a0[cc])[1] = *(const v8h*)(zr0 + koff + 8);
        ((v8h*)&a1[cc])[0] = *(const v8h*)(zr1 + koff);
        ((v8h*)&a1[cc])[1] = *(const v8h*)(zr1 + koff + 8);
    }

    const unsigned smb = (unsigned)(size_t)(&smem[0]);       // LDS byte base
    const char*    cns = (const char*)(cn + (size_t)cbase * VQ_C);

    // Stage one 32-code tile (16 KiB) into padded LDS buffer (ktile % 3).
    // 128 threads x 8 insts x 16B; each wave issues 8 async instructions.
    auto stage = [&](int ktile) {
        const unsigned bufb = smb + (unsigned)((ktile % 3) * TILE_PAD_BYTES);
        const char* src = cns + (size_t)ktile * TILE_SRC_BYTES;
        #pragma unroll
        for (int inst = 0; inst < 8; ++inst) {
            const int ib  = t * 16 + inst * 2048;            // linear byte
            const int row = ib >> 9;
            const int off = ib & 511;
            const unsigned lds = bufb + (unsigned)(row * 528 + off);
            const char* g = src + ib;
            asm volatile("global_load_async_to_lds_b128 %0, %1, off"
                         :: "v"(lds), "v"(g) : "memory");
        }
    };

    float best0[8], best1[8];
    int   bi0[8],   bi1[8];
    #pragma unroll
    for (int v = 0; v < 8; ++v) {
        best0[v] = -3.4e38f; best1[v] = -3.4e38f;
        bi0[v] = 0; bi1[v] = 0;
    }

    stage(0);
    stage(1);

    for (int kt = 0; kt < KT_PER; ++kt) {
        if (kt < KT_PER - 1)
            asm volatile("s_wait_asynccnt 0x8" ::: "memory");  // tile kt landed
        else
            asm volatile("s_wait_asynccnt 0x0" ::: "memory");
        __syncthreads();          // tile kt visible; buffer (kt+2)%3 now free
        if (kt + 2 < KT_PER) stage(kt + 2);

        const _Float16* bt  = &smem[(kt % 3) * TILE_PAD_HALFS];
        const _Float16* bp0 = bt + n15 * LDSP + hi * 16;           // col grp 0
        const _Float16* bp1 = bt + (16 + n15) * LDSP + hi * 16;    // col grp 1

        v8f acc00 = {}, acc01 = {}, acc10 = {}, acc11 = {};
        v16h bc0, bc1, bn0, bn1;
        ((v8h*)&bc0)[0] = *(const v8h*)(bp0);
        ((v8h*)&bc0)[1] = *(const v8h*)(bp0 + 8);
        ((v8h*)&bc1)[0] = *(const v8h*)(bp1);
        ((v8h*)&bc1)[1] = *(const v8h*)(bp1 + 8);
        #pragma unroll
        for (int cc = 0; cc < 8; ++cc) {
            if (cc < 7) {            // prefetch next chunk while computing
                const int o = (cc + 1) * 32;
                ((v8h*)&bn0)[0] = *(const v8h*)(bp0 + o);
                ((v8h*)&bn0)[1] = *(const v8h*)(bp0 + o + 8);
                ((v8h*)&bn1)[0] = *(const v8h*)(bp1 + o);
                ((v8h*)&bn1)[1] = *(const v8h*)(bp1 + o + 8);
            }
            acc00 = __builtin_amdgcn_wmma_f32_16x16x32_f16(
                        false, a0[cc], false, bc0, (short)0, acc00, false, false);
            acc10 = __builtin_amdgcn_wmma_f32_16x16x32_f16(
                        false, a1[cc], false, bc0, (short)0, acc10, false, false);
            acc01 = __builtin_amdgcn_wmma_f32_16x16x32_f16(
                        false, a0[cc], false, bc1, (short)0, acc01, false, false);
            acc11 = __builtin_amdgcn_wmma_f32_16x16x32_f16(
                        false, a1[cc], false, bc1, (short)0, acc11, false, false);
            bc0 = bn0;
            bc1 = bn1;
        }

        const int code0 = cbase + kt * TCODES + n15;   // column group 0
        const int code1 = code0 + 16;                  // column group 1
        #pragma unroll
        for (int v = 0; v < 8; ++v) {
            float x = acc00[v];
            bool  g = x > best0[v];
            best0[v] = g ? x : best0[v];
            bi0[v]   = g ? code0 : bi0[v];
            x = acc01[v];                 // higher codes second: ties keep grp0
            g = x > best0[v];
            best0[v] = g ? x : best0[v];
            bi0[v]   = g ? code1 : bi0[v];
            x = acc10[v];
            g = x > best1[v];
            best1[v] = g ? x : best1[v];
            bi1[v]   = g ? code0 : bi1[v];
            x = acc11[v];
            g = x > best1[v];
            best1[v] = g ? x : best1[v];
            bi1[v]   = g ? code1 : bi1[v];
        }
    }

    // Cross-lane argmax over the 16 columns (stays within 16-lane halves).
    #pragma unroll
    for (int m = 1; m <= 8; m <<= 1) {
        #pragma unroll
        for (int v = 0; v < 8; ++v) {
            float ov = __shfl_xor(best0[v], m, 32);
            int   oi = __shfl_xor(bi0[v],   m, 32);
            bool take = (ov > best0[v]) || (ov == best0[v] && oi < bi0[v]);
            best0[v] = take ? ov : best0[v];
            bi0[v]   = take ? oi : bi0[v];
            ov = __shfl_xor(best1[v], m, 32);
            oi = __shfl_xor(bi1[v],   m, 32);
            take = (ov > best1[v]) || (ov == best1[v] && oi < bi1[v]);
            best1[v] = take ? ov : best1[v];
            bi1[v]   = take ? oi : bi1[v];
        }
    }

    // D layout: VGPR v -> row v (lanes 0-15) / row v+8 (lanes 16-31).
    #pragma unroll
    for (int v = 0; v < 8; ++v) {
        if (lane == v) {
            pbest[(long)(mbase + v)      * KSPLIT + ks] = best0[v];
            pidx [(long)(mbase + v)      * KSPLIT + ks] = bi0[v];
            pbest[(long)(mbase + 16 + v) * KSPLIT + ks] = best1[v];
            pidx [(long)(mbase + 16 + v) * KSPLIT + ks] = bi1[v];
        }
        if (lane == 16 + v) {
            pbest[(long)(mbase + 8 + v)  * KSPLIT + ks] = best0[v];
            pidx [(long)(mbase + 8 + v)  * KSPLIT + ks] = bi0[v];
            pbest[(long)(mbase + 24 + v) * KSPLIT + ks] = best1[v];
            pidx [(long)(mbase + 24 + v) * KSPLIT + ks] = bi1[v];
        }
    }
}

// ---------------------------------------------------------------------------
// Kernel 3b: combine K-slice partials (ascending slice order => argmax ties
// resolve to the lowest code index, matching jnp.argmax).
// ---------------------------------------------------------------------------
__global__ void vq_argmax_reduce(const float* __restrict__ pbest,
                                 const int* __restrict__ pidx,
                                 int* __restrict__ idx,
                                 float* __restrict__ fidx) {
    const int n = blockIdx.x * blockDim.x + threadIdx.x;
    float b = -3.4e38f;
    int   bi = 0;
    #pragma unroll
    for (int s = 0; s < KSPLIT; ++s) {
        const float v = pbest[(long)n * KSPLIT + s];
        const int   i = pidx [(long)n * KSPLIT + s];
        const bool tk = v > b;
        b  = tk ? v : b;
        bi = tk ? i : bi;
    }
    idx[n]  = bi;
    fidx[n] = (float)bi;
}

// ---------------------------------------------------------------------------
// Kernel 4: gather z_q = embed[idx], write output [B,C,H,W], loss partials.
// ---------------------------------------------------------------------------
__global__ void vq_gather_out_loss(const float* __restrict__ z,
                                   const float* __restrict__ embed,
                                   const int* __restrict__ idx,
                                   float* __restrict__ out,
                                   float* __restrict__ partial) {
    const int bh = blockIdx.x;
    const int b  = bh >> 5;
    const int h  = bh & 31;
    const int t  = threadIdx.x;
    const int w  = t & 31;
    const int cg = t >> 5;
    const float* zb = z   + (((long)b * VQ_C) * VQ_H + h) * VQ_W;
    float*       ob = out + (((long)b * VQ_C) * VQ_H + h) * VQ_W;
    const int id = idx[bh * VQ_W + w];
    const float* er = embed + (long)id * VQ_C;
    float ls = 0.0f;
    #pragma unroll 4
    for (int cb = 0; cb < VQ_C; cb += 8) {
        const int c = cb + cg;
        const float e  = er[c];
        const float zv = zb[(long)c * VQ_HW + w];
        ob[(long)c * VQ_HW + w] = e;
        const float d = e - zv;
        ls += d * d;
    }
    __shared__ float red[256];
    red[t] = ls;
    __syncthreads();
    #pragma unroll
    for (int s = 128; s > 0; s >>= 1) {
        if (t < s) red[t] += red[t + s];
        __syncthreads();
    }
    if (t == 0) partial[bh] = red[0];
}

// ---------------------------------------------------------------------------
// Kernel 5: deterministic final loss:  1.25 * mean((z_q - z)^2).
// ---------------------------------------------------------------------------
__global__ void vq_loss_final(const float* __restrict__ partial,
                              float* __restrict__ loss) {
    const int t = threadIdx.x;
    __shared__ float red[256];
    float s = 0.0f;
    for (int i = t; i < VQ_B * VQ_H; i += 256) s += partial[i];
    red[t] = s;
    __syncthreads();
    #pragma unroll
    for (int st = 128; st > 0; st >>= 1) {
        if (t < st) red[t] += red[t + st];
        __syncthreads();
    }
    if (t == 0) loss[0] = red[0] * 1.25f / (float)VQ_OUT_ELEMS;
}

// ---------------------------------------------------------------------------
extern "C" void kernel_launch(void* const* d_in, const int* in_sizes, int n_in,
                              void* d_out, int out_size, void* d_ws, size_t ws_size,
                              hipStream_t stream) {
    const float* z     = (const float*)d_in[0];   // [32,256,32,32]
    const float* embed = (const float*)d_in[1];   // [8192,256]

    float* out  = (float*)d_out;                  // 8388608 floats
    float* loss = out + VQ_OUT_ELEMS;             // 1 float
    float* fidx = out + VQ_OUT_ELEMS + 1;         // 32768 floats (indices)

    char* ws = (char*)d_ws;
    size_t o = 0;
    _Float16* zn      = (_Float16*)(ws + o); o += (size_t)VQ_N * VQ_C * 2;  // 16 MiB
    _Float16* cn      = (_Float16*)(ws + o); o += (size_t)VQ_K * VQ_C * 2;  //  4 MiB
    int*      idx     = (int*)     (ws + o); o += (size_t)VQ_N * 4;         // 128 KiB
    float*    partial = (float*)   (ws + o); o += (size_t)VQ_B * VQ_H * 4;  //   4 KiB
    float*    pbest   = (float*)   (ws + o); o += (size_t)VQ_N * KSPLIT * 4;//   1 MiB
    int*      pidx    = (int*)     (ws + o); o += (size_t)VQ_N * KSPLIT * 4;//   1 MiB

    vq_norm_embed<<<VQ_K, VQ_C, 0, stream>>>(embed, cn);
    vq_norm_z<<<VQ_B * VQ_H, 256, 0, stream>>>(z, zn);
    vq_dist_argmax<<<NBLK_ROWS * KSPLIT, ROWS_PB, 0, stream>>>(zn, cn, pbest, pidx);
    vq_argmax_reduce<<<VQ_N / 256, 256, 0, stream>>>(pbest, pidx, idx, fidx);
    vq_gather_out_loss<<<VQ_B * VQ_H, 256, 0, stream>>>(z, embed, idx, out, partial);
    vq_loss_final<<<1, 256, 0, stream>>>(partial, loss);
}
